// TransformerEncoder_28853590294724
// MI455X (gfx1250) — compile-verified, matrix operand order checked by reading, not run on hardware
//
#include <hip/hip_runtime.h>
#include <hip/hip_bf16.h>
#include <math.h>

typedef _Float16 f16;
typedef _Float16 v16h __attribute__((ext_vector_type(16)));
typedef _Float16 v8h  __attribute__((ext_vector_type(8)));
typedef _Float16 v4h  __attribute__((ext_vector_type(4)));
typedef float    v8f  __attribute__((ext_vector_type(8)));
typedef float    v4f  __attribute__((ext_vector_type(4)));
typedef int      i32x4 __attribute__((vector_size(16)));

#define IDIM 256
#define NU   512
#define EU   2048
#define NL   4
#define NH   8
#define DK   64
#define TT   1024
#define NB   8
#define MTOK 8192   /* NB * TT */

#define WMMA(a,b,c) __builtin_amdgcn_wmma_f32_16x16x32_f16(false,(a),false,(b),(short)0,(c),false,false)

// gfx1250 async global->LDS path (ASYNCcnt) if the toolchain exposes it.
#if defined(__has_builtin)
#if __has_builtin(__builtin_amdgcn_global_load_async_to_lds_b128) && \
    __has_builtin(__builtin_amdgcn_s_wait_asynccnt)
#define USE_ASYNC_LDS 1
#endif
#endif
#ifndef USE_ASYNC_LDS
#define USE_ASYNC_LDS 0
#endif

// Copy 16 bytes (8 halves) global -> LDS, per lane.
__device__ __forceinline__ void stage16(const f16* g, f16* l) {
#if USE_ASYNC_LDS
  __builtin_amdgcn_global_load_async_to_lds_b128(
      (__attribute__((address_space(1))) i32x4*)g,
      (__attribute__((address_space(3))) i32x4*)l, 0, 0);
#else
  *(v8h*)l = *(const v8h*)g;
#endif
}

__device__ __forceinline__ void stage_drain() {
#if USE_ASYNC_LDS
  __builtin_amdgcn_s_wait_asynccnt(0);
#endif
}

// Load one 16-lane x 32-K half-precision fragment: per lane two contiguous
// 16-byte chunks at +0 and +16 halves (matches CDNA5 16-bit A/B VGPR layout
// when the caller applies the (lane<16 ? 0 : 8) k-base offset).
__device__ __forceinline__ v16h load_frag(const f16* p) {
  v8h lo = *(const v8h*)(p);
  v8h hi = *(const v8h*)(p + 16);
  v16h r;
#pragma unroll
  for (int i = 0; i < 8; ++i) { r[i] = lo[i]; r[i + 8] = hi[i]; }
  return r;
}

// ---------------------------------------------------------------- cvt f32->f16
__global__ __launch_bounds__(256)
void cvt_f32_f16(const float* __restrict__ x, f16* __restrict__ y, int n) {
  int i = (blockIdx.x * 256 + threadIdx.x) * 4;
  if (i < n) {
    v4f v = *(const v4f*)(x + i);
    v4h o;
#pragma unroll
    for (int j = 0; j < 4; ++j) o[j] = (f16)v[j];
    *(v4h*)(y + i) = o;
  }
}

// ------------------------------------------------------------------- layernorm
// one block per row of 512; in-place safe (x may == y)
__global__ __launch_bounds__(256)
void layernorm(const float* x, const float* __restrict__ g,
               const float* __restrict__ b, float* y, f16* __restrict__ y16) {
  const int row = blockIdx.x;
  const int tid = threadIdx.x;
  const float* xr = x + (size_t)row * NU;
  float2 v = *(const float2*)(xr + tid * 2);

  __shared__ float red[8];
  __shared__ float red2[8];
  const int wid = tid >> 5;

  float s = v.x + v.y;
#pragma unroll
  for (int m = 1; m <= 16; m <<= 1) s += __shfl_xor(s, m);
  if ((tid & 31) == 0) red[wid] = s;
  __syncthreads();
  float tot = 0.f;
#pragma unroll
  for (int i = 0; i < 8; ++i) tot += red[i];
  const float mu = tot * (1.0f / NU);

  const float dx = v.x - mu, dy = v.y - mu;
  float vs = dx * dx + dy * dy;
#pragma unroll
  for (int m = 1; m <= 16; m <<= 1) vs += __shfl_xor(vs, m);
  if ((tid & 31) == 0) red2[wid] = vs;
  __syncthreads();
  float vt = 0.f;
#pragma unroll
  for (int i = 0; i < 8; ++i) vt += red2[i];
  const float rstd = rsqrtf(vt * (1.0f / NU) + 1e-5f);

  const int c0 = tid * 2;
  const float o0 = dx * rstd * g[c0] + b[c0];
  const float o1 = dy * rstd * g[c0 + 1] + b[c0 + 1];
  size_t o = (size_t)row * NU + c0;
  y[o] = o0;
  y[o + 1] = o1;
  if (y16) { y16[o] = (f16)o0; y16[o + 1] = (f16)o1; }
}

// ------------------------------------------------------------------ WMMA GEMM
// C[m,n] = sum_k A16[m,k] * B16[n,k] (+bias) (+resid) (ReLU?)  -> f32 / f16
// flags: 1=relu 2=resid 4=write f32 8=write f16
__global__ __launch_bounds__(32)
void gemm_wmma(const f16* __restrict__ A, const f16* __restrict__ B,
               const float* __restrict__ bias, const float* __restrict__ resid,
               float* __restrict__ C32, f16* __restrict__ C16,
               int M, int N, int K, int flags) {
  const int lane = threadIdx.x;
  const int hf = lane >> 4;
  const int r = lane & 15;
  const int kb = hf << 3;
  const int n0 = blockIdx.x << 5;
  const int m0 = blockIdx.y << 5;

  const f16* a0 = A + (size_t)(m0 + r) * K + kb;
  const f16* a1 = A + (size_t)(m0 + 16 + r) * K + kb;
  const f16* b0 = B + (size_t)(n0 + r) * K + kb;
  const f16* b1 = B + (size_t)(n0 + 16 + r) * K + kb;

  v8f c00 = {}, c01 = {}, c10 = {}, c11 = {};
  for (int kk = 0; kk < K; kk += 32) {
    if (kk + 128 < K) {
      __builtin_prefetch(a0 + kk + 128);
      __builtin_prefetch(b0 + kk + 128);
    }
    v16h af0 = load_frag(a0 + kk);
    v16h af1 = load_frag(a1 + kk);
    v16h bf0 = load_frag(b0 + kk);
    v16h bf1 = load_frag(b1 + kk);
    c00 = WMMA(af0, bf0, c00);
    c01 = WMMA(af0, bf1, c01);
    c10 = WMMA(af1, bf0, c10);
    c11 = WMMA(af1, bf1, c11);
  }

  const bool relu = flags & 1, hasres = flags & 2, wf32 = flags & 4, wf16 = flags & 8;
  v8f cc[2][2] = {{c00, c01}, {c10, c11}};
#pragma unroll
  for (int mi = 0; mi < 2; ++mi)
#pragma unroll
    for (int ni = 0; ni < 2; ++ni) {
      const int col = n0 + ni * 16 + r;
      const float bv = bias[col];
#pragma unroll
      for (int i = 0; i < 8; ++i) {
        const size_t row = (size_t)(m0 + mi * 16 + hf * 8 + i);
        float v = cc[mi][ni][i] + bv;
        if (hasres) v += resid[row * N + col];
        if (relu) v = fmaxf(v, 0.f);
        if (wf32) C32[row * N + col] = v;
        if (wf16) C16[row * N + col] = (f16)v;
      }
    }
}

// ----------------------------------------------------------- flash attention
// grid = (T/64, NH, NB), block = 128 (4 waves, 16 q-rows each)
__global__ __launch_bounds__(128)
void attention(const f16* __restrict__ Q, const f16* __restrict__ Kh,
               const f16* __restrict__ V, f16* __restrict__ ctx16) {
  __shared__ f16 lds_k[32 * 72];        // K tile [s=32][d=64] (+pad)
  __shared__ f16 lds_vt[64 * 40];       // V^T tile [d=64][s=32] (+pad)
  __shared__ f16 lds_p[4][16 * 40];     // per-wave P [m=16][s=32] (+pad)

  const int tid = threadIdx.x;
  const int wave = tid >> 5;
  const int lane = tid & 31;
  const int hf = lane >> 4;
  const int r = lane & 15;
  const int kb = hf << 3;

  const int m0 = blockIdx.x * 64 + wave * 16;   // q-row base within T
  const int h = blockIdx.y;
  const int bi = blockIdx.z;
  const size_t base = (size_t)bi * TT * NU + h * DK;

  // Q fragments (16 rows x 64 k, two 16x32 frags), resident for whole loop
  const f16* qp = Q + base + (size_t)(m0 + r) * NU;
  const v16h qa0 = load_frag(qp + kb);
  const v16h qa1 = load_frag(qp + 32 + kb);

  v8f o[4] = {v8f{}, v8f{}, v8f{}, v8f{}};
  float mx[8], ls[8];
#pragma unroll
  for (int i = 0; i < 8; ++i) { mx[i] = -3.0e38f; ls[i] = 0.f; }
  const float scale = 0.125f;           // 1/sqrt(64)

  const int sld = tid >> 2;             // kv row within tile, 0..31
  const int dseg = (tid & 3) * 16;      // 16-wide d segment

  for (int j = 0; j < TT / 32; ++j) {
    __syncthreads();
    { // stage K tile via CDNA5 async global->LDS; V transposed via VGPRs
      const f16* kp = Kh + base + (size_t)(j * 32 + sld) * NU + dseg;
      stage16(kp, &lds_k[sld * 72 + dseg]);
      stage16(kp + 8, &lds_k[sld * 72 + dseg + 8]);
      const f16* vp = V + base + (size_t)(j * 32 + sld) * NU + dseg;
      v8h v0 = *(const v8h*)vp;
      v8h v1 = *(const v8h*)(vp + 8);
#pragma unroll
      for (int i = 0; i < 8; ++i) {
        lds_vt[(dseg + i) * 40 + sld] = v0[i];
        lds_vt[(dseg + 8 + i) * 40 + sld] = v1[i];
      }
      stage_drain();                    // ASYNCcnt -> 0 before publishing tile
    }
    __syncthreads();

    // S[16x32] = Q * K^T   (K tile rows are B-operand columns)
    v8f s0 = {}, s1 = {};
    {
      const f16* kr0 = &lds_k[r * 72];
      const f16* kr1 = &lds_k[(16 + r) * 72];
      s0 = WMMA(qa0, load_frag(kr0 + kb), s0);
      s0 = WMMA(qa1, load_frag(kr0 + 32 + kb), s0);
      s1 = WMMA(qa0, load_frag(kr1 + kb), s1);
      s1 = WMMA(qa1, load_frag(kr1 + 32 + kb), s1);
    }

    // online softmax; per-row stats live in the C-layout lane mapping
    float p0[8], p1[8];
#pragma unroll
    for (int i = 0; i < 8; ++i) {
      const float a = s0[i] * scale, c = s1[i] * scale;
      float t = fmaxf(a, c);
      t = fmaxf(t, __shfl_xor(t, 1));
      t = fmaxf(t, __shfl_xor(t, 2));
      t = fmaxf(t, __shfl_xor(t, 4));
      t = fmaxf(t, __shfl_xor(t, 8));
      const float nm = fmaxf(mx[i], t);
      const float al = __expf(mx[i] - nm);
      mx[i] = nm;
      p0[i] = __expf(a - nm);
      p1[i] = __expf(c - nm);
      float rs = p0[i] + p1[i];
      rs += __shfl_xor(rs, 1);
      rs += __shfl_xor(rs, 2);
      rs += __shfl_xor(rs, 4);
      rs += __shfl_xor(rs, 8);
      ls[i] = ls[i] * al + rs;
      o[0][i] *= al; o[1][i] *= al; o[2][i] *= al; o[3][i] *= al;
    }

    // relayout P (C-layout f32 -> A-layout f16) through per-wave LDS
    f16* pw = &lds_p[wave][0];
#pragma unroll
    for (int i = 0; i < 8; ++i) {
      pw[(hf * 8 + i) * 40 + r] = (f16)p0[i];
      pw[(hf * 8 + i) * 40 + 16 + r] = (f16)p1[i];
    }
    __syncthreads();

    const v16h pa = load_frag(pw + r * 40 + kb);
#pragma unroll
    for (int nt = 0; nt < 4; ++nt) {
      const v16h vb = load_frag(&lds_vt[(nt * 16 + r) * 40 + kb]);
      o[nt] = WMMA(pa, vb, o[nt]);
    }
  }

  // finalize: divide by row sums and write ctx (f16 feed for Wo GEMM)
#pragma unroll
  for (int i = 0; i < 8; ++i) {
    const float inv = 1.0f / ls[i];
    const size_t row = (size_t)bi * TT + m0 + hf * 8 + i;
#pragma unroll
    for (int nt = 0; nt < 4; ++nt)
      ctx16[row * NU + h * DK + nt * 16 + r] = (f16)(o[nt][i] * inv);
  }
}

// ------------------------------------------------------------------- launcher
extern "C" void kernel_launch(void* const* d_in, const int* in_sizes, int n_in,
                              void* d_out, int out_size, void* d_ws, size_t ws_size,
                              hipStream_t stream) {
  (void)in_sizes; (void)n_in; (void)out_size; (void)ws_size;
  const float* x    = (const float*)d_in[0];
  const float* Win  = (const float*)d_in[1];
  const float* bin_ = (const float*)d_in[2];
  const float* ln1g = (const float*)d_in[3];
  const float* ln1b = (const float*)d_in[4];
  const float* Wq   = (const float*)d_in[5];
  const float* bq   = (const float*)d_in[6];
  const float* Wk   = (const float*)d_in[7];
  const float* bk   = (const float*)d_in[8];
  const float* Wv   = (const float*)d_in[9];
  const float* bv   = (const float*)d_in[10];
  const float* Wo   = (const float*)d_in[11];
  const float* bo   = (const float*)d_in[12];
  const float* ln2g = (const float*)d_in[13];
  const float* ln2b = (const float*)d_in[14];
  const float* W1   = (const float*)d_in[15];
  const float* b1   = (const float*)d_in[16];
  const float* W2   = (const float*)d_in[17];
  const float* b2   = (const float*)d_in[18];
  const float* lnog = (const float*)d_in[19];
  const float* lnob = (const float*)d_in[20];

  char* p = (char*)d_ws;
  auto carve = [&](size_t bytes) {
    char* q = p;
    p += (bytes + 255) & ~(size_t)255;
    return (void*)q;
  };
  f16* x16   = (f16*)carve((size_t)MTOK * IDIM * 2);
  f16* Win16 = (f16*)carve((size_t)NU * IDIM * 2);
  f16* Wq16  = (f16*)carve((size_t)NL * NU * NU * 2);
  f16* Wk16  = (f16*)carve((size_t)NL * NU * NU * 2);
  f16* Wv16  = (f16*)carve((size_t)NL * NU * NU * 2);
  f16* Wo16  = (f16*)carve((size_t)NL * NU * NU * 2);
  f16* W116  = (f16*)carve((size_t)NL * EU * NU * 2);
  f16* W216  = (f16*)carve((size_t)NL * NU * EU * 2);
  float* e   = (float*)carve((size_t)MTOK * NU * 4);
  f16* e16   = (f16*)carve((size_t)MTOK * NU * 2);
  f16* q16   = (f16*)carve((size_t)MTOK * NU * 2);
  f16* k16   = (f16*)carve((size_t)MTOK * NU * 2);
  f16* v16   = (f16*)carve((size_t)MTOK * NU * 2);
  f16* c16   = (f16*)carve((size_t)MTOK * NU * 2);
  f16* h16   = (f16*)carve((size_t)MTOK * EU * 2);

  auto cvt = [&](const float* src, f16* dst, int n) {
    cvt_f32_f16<<<dim3((n / 4 + 255) / 256), dim3(256), 0, stream>>>(src, dst, n);
  };
  cvt(x, x16, MTOK * IDIM);
  cvt(Win, Win16, NU * IDIM);
  cvt(Wq, Wq16, NL * NU * NU);
  cvt(Wk, Wk16, NL * NU * NU);
  cvt(Wv, Wv16, NL * NU * NU);
  cvt(Wo, Wo16, NL * NU * NU);
  cvt(W1, W116, NL * EU * NU);
  cvt(W2, W216, NL * NU * EU);

  auto gemm = [&](const f16* A, const f16* B, const float* bias,
                  const float* resid, float* C32, f16* C16,
                  int M, int N, int K, int flags) {
    gemm_wmma<<<dim3(N / 32, M / 32), dim3(32), 0, stream>>>(
        A, B, bias, resid, C32, C16, M, N, K, flags);
  };

  // input projection: e = x @ Win^T + bin  (f32 residual stream)
  gemm(x16, Win16, bin_, nullptr, e, nullptr, MTOK, NU, IDIM, 4);

  for (int l = 0; l < NL; ++l) {
    // e = LN1(e) (in place) + f16 copy
    layernorm<<<dim3(MTOK), dim3(256), 0, stream>>>(e, ln1g + l * NU, ln1b + l * NU, e, e16);
    // q/k/v projections (f16 outputs only)
    gemm(e16, Wq16 + (size_t)l * NU * NU, bq + l * NU, nullptr, nullptr, q16, MTOK, NU, NU, 8);
    gemm(e16, Wk16 + (size_t)l * NU * NU, bk + l * NU, nullptr, nullptr, k16, MTOK, NU, NU, 8);
    gemm(e16, Wv16 + (size_t)l * NU * NU, bv + l * NU, nullptr, nullptr, v16, MTOK, NU, NU, 8);
    // flash attention -> ctx f16
    attention<<<dim3(TT / 64, NH, NB), dim3(128), 0, stream>>>(q16, k16, v16, c16);
    // e = e + ctx @ Wo^T + bo
    gemm(c16, Wo16 + (size_t)l * NU * NU, bo + l * NU, e, e, nullptr, MTOK, NU, NU, 2 | 4);
    // e = LN2(e) (in place) + f16 copy
    layernorm<<<dim3(MTOK), dim3(256), 0, stream>>>(e, ln2g + l * NU, ln2b + l * NU, e, e16);
    // hdn = relu(e @ W1^T + b1) (f16 only)
    gemm(e16, W116 + (size_t)l * EU * NU, b1 + l * EU, nullptr, nullptr, h16, MTOK, EU, NU, 1 | 8);
    // e = e + hdn @ W2^T + b2
    gemm(h16, W216 + (size_t)l * NU * EU, b2 + l * NU, e, e, nullptr, MTOK, NU, EU, 2 | 4);
  }

  // out = LN(e) -> d_out (fp32)
  layernorm<<<dim3(MTOK), dim3(256), 0, stream>>>(e, lnog, lnob, (float*)d_out, nullptr);
}
